// NeuronConvNet_23347442221453
// MI455X (gfx1250) — matrix operand order repaired and checked by reading, not compiled, not run on hardware
//
#include <hip/hip_runtime.h>
#include <hip/hip_bf16.h>

typedef __attribute__((ext_vector_type(16))) __bf16 v16bf;
typedef __attribute__((ext_vector_type(8)))  __bf16 v8bf;
typedef __attribute__((ext_vector_type(8)))  float  v8f;

#define T_LEN 4096
#define TILE_T 128
#define KTAPS 15
#define HALO 7
#define NEG_SLOPE 0.01f

__device__ __forceinline__ __bf16 f2bf(float f) {
    union { float f; unsigned u; } in;
    in.f = f;
    unsigned r = in.u + 0x7FFFu + ((in.u >> 16) & 1u);   // round-to-nearest-even
    union { unsigned short s; __bf16 b; } out;
    out.s = (unsigned short)(r >> 16);
    return out.b;
}

__device__ __forceinline__ __bf16 to_bf(float v)  { return f2bf(v); }
__device__ __forceinline__ __bf16 to_bf(__bf16 v) { return v; }

__device__ __forceinline__ void store_act(float*  p, float v) { *p = v; }
__device__ __forceinline__ void store_act(__bf16* p, float v) { *p = f2bf(v); }

// Grouped 1-D conv (K=15, SAME) as tap-decomposed implicit GEMM on
// v_wmma_f32_16x16x32_bf16.  WMMA K-dim = input channels (channel-major LDS
// tiles), so every A/B fragment is built from contiguous ds_load_b128s.
//   I==16 layers: two taps per WMMA (K=32 fully used), 8 steps.
//   leaf (I=20):  one tap per WMMA, channels padded to 32, 15 steps.
// 8 waves per WG, each wave owns a 16-time x 16-outch D tile (128 times/WG).
// Intermediates live in HBM as bf16 (half the traffic of f32); raw input x
// and the final output stay f32 per the reference dtypes.
template<int IPG0, int IPG1, int OPG, bool ACT,
         typename TI0, typename TI1, typename TO>
__global__ __launch_bounds__(256)
void gconv_wmma(const TI0* __restrict__ in0,
                const TI1* __restrict__ in1,
                const float* __restrict__ W,
                const float* __restrict__ bias,
                TO* __restrict__ out,
                int G, int in0_nch, int in0_choff, int in1_nch, int in1_choff)
{
    constexpr int  IPG   = IPG0 + IPG1;
    constexpr bool PAIR  = (IPG <= 16);          // two taps per WMMA step
    constexpr int  CHP   = PAIR ? 16 : 32;       // padded channel count
    constexpr int  NSTEP = PAIR ? 8 : 15;
    constexpr int  NTAP  = PAIR ? 16 : 15;       // allocated taps (15 real)
    constexpr int  XVAL  = TILE_T + 2 * HALO;    // 142 valid halo samples
    constexpr int  XTP   = 144;                  // padded rows

    __shared__ __bf16 sXt[XTP][CHP];             // [time][channel], zero padded
    __shared__ __bf16 sWt[NTAP][16][CHP];        // [tap][outch][channel], padded

    const int tid = threadIdx.x;
    const int g   = blockIdx.y;
    const int b   = blockIdx.z;
    const int t0  = blockIdx.x * TILE_T;

    // ---- stage weights: W[o][i][k] -> sWt[k][n][i] (bf16, zero padded) ----
    for (int idx = tid; idx < NTAP * 16 * CHP; idx += 256) {
        int tap = idx / (16 * CHP);
        int rem = idx - tap * (16 * CHP);
        int n   = rem / CHP;
        int ch  = rem - n * CHP;
        float w = 0.0f;
        if (tap < KTAPS && n < OPG && ch < IPG)
            w = W[((size_t)(g * OPG + n) * IPG + ch) * KTAPS + tap];
        sWt[tap][n][ch] = f2bf(w);
    }
    // ---- stage input halo tile: channel-major rows, coalesced global reads --
    for (int idx = tid; idx < CHP * XTP; idx += 256) {
        int ch = idx / XTP;                      // channel (slow) -> t coalesced
        int tl = idx - ch * XTP;
        int t  = t0 - HALO + tl;
        __bf16 v = f2bf(0.0f);
        if (ch < IPG && tl < XVAL && t >= 0 && t < T_LEN) {
            if (IPG1 == 0 || ch < IPG0)
                v = to_bf(in0[((size_t)b * in0_nch + in0_choff + g * IPG0 + ch) * T_LEN + t]);
            else
                v = to_bf(in1[((size_t)b * in1_nch + in1_choff + g * IPG1 + (ch - IPG0)) * T_LEN + t]);
        }
        sXt[tl][ch] = v;
    }
    __syncthreads();

    const int lane = tid & 31;
    const int wave = tid >> 5;
    const int half = lane >> 4;
    const int l15  = lane & 15;
    const int mrow = wave * 16 + l15;            // time row for this lane's A

    v8f c = {};
#pragma unroll
    for (int s = 0; s < NSTEP; ++s) {
        v16bf a, bfrag;
        if constexpr (PAIR) {
            // K = [tap 2s: ch0-15 | tap 2s+1: ch0-15]
            // A: e <-> K=e+16h -> tap 2s+h, ch e (32B contiguous)
            a = *(const v16bf*)&sXt[mrow + 2 * s + half][0];
            // B: e <-> K=e+16h -> tap 2s+h, ch e (32B contiguous)
            bfrag = *(const v16bf*)&sWt[2 * s + half][l15][0];
        } else {
            // K = 32 channels (20 real) of tap s
            a = *(const v16bf*)&sXt[mrow + s][16 * half];
            bfrag = *(const v16bf*)&sWt[s][l15][16 * half];
        }
        c = __builtin_amdgcn_wmma_f32_16x16x32_bf16(false, a, false, bfrag,
                                                    (short)0, c, false, false);
    }

    // ---- epilogue: bias + LeakyReLU.  D: VGPR r <-> M=r+8*half.
    const int n = l15;
    if (n < OPG) {
        const float bv = bias[g * OPG + n];
        TO* op = out + (((size_t)b * G + g) * OPG + n) * T_LEN + t0 + wave * 16;
#pragma unroll
        for (int r = 0; r < 8; ++r) {
            float v = c[r] + bv;
            if (ACT) v = (v > 0.0f) ? v : v * NEG_SLOPE;
            store_act(op + r + 8 * half, v);
        }
    }
}

extern "C" void kernel_launch(void* const* d_in, const int* in_sizes, int n_in,
                              void* d_out, int out_size, void* d_ws, size_t ws_size,
                              hipStream_t stream)
{
    const float* x      = (const float*)d_in[0];
    const float* W_leaf = (const float*)d_in[1];
    const float* b_leaf = (const float*)d_in[2];
    const float* W_int0 = (const float*)d_in[3];
    const float* b_int0 = (const float*)d_in[4];
    const float* W_br   = (const float*)d_in[5];
    const float* b_br   = (const float*)d_in[6];
    const float* W_int1 = (const float*)d_in[7];
    const float* b_int1 = (const float*)d_in[8];
    const float* W_int2 = (const float*)d_in[9];
    const float* b_int2 = (const float*)d_in[10];
    const float* W_int3 = (const float*)d_in[11];
    const float* b_int3 = (const float*)d_in[12];
    const float* W_int4 = (const float*)d_in[13];
    const float* b_int4 = (const float*)d_in[14];
    const float* W_root = (const float*)d_in[15];
    const float* b_root = (const float*)d_in[16];

    // bf16 ping-pong scratch: A holds up to 512 channels, B up to 256 channels
    __bf16* bufA = (__bf16*)d_ws;
    __bf16* bufB = (__bf16*)((char*)d_ws + (size_t)16 * 512 * T_LEN * sizeof(__bf16));
    const __bf16* nobf = nullptr;

    const dim3 blk(256);
    const int NT = T_LEN / TILE_T;   // 32 time tiles

    // leaf : x[:,0:1280] (f32)     -> A (B,512,T) bf16, groups=64, I=20
    gconv_wmma<20, 0, 8, true , float,  float,  __bf16><<<dim3(NT, 64, 16), blk, 0, stream>>>(
        x,    (const float*)nullptr, W_leaf, b_leaf, bufA, 64, 1536, 0,    0,    0);
    // int0 : A (512)               -> B (B,256,T) bf16, groups=32, I=16
    gconv_wmma<16, 0, 8, true , __bf16, __bf16, __bf16><<<dim3(NT, 32, 16), blk, 0, stream>>>(
        bufA, nobf, W_int0, b_int0, bufB, 32,  512, 0,    0,    0);
    // br   : concat(B(256) bf16, x[:,1280:] f32) -> A (B,256,T) bf16, groups=32
    gconv_wmma< 8, 8, 8, true , __bf16, float,  __bf16><<<dim3(NT, 32, 16), blk, 0, stream>>>(
        bufB, x,    W_br,   b_br,   bufA, 32,  256, 0, 1536, 1280);
    // int1 : A (256) -> B (B,128,T), groups=16
    gconv_wmma<16, 0, 8, true , __bf16, __bf16, __bf16><<<dim3(NT, 16, 16), blk, 0, stream>>>(
        bufA, nobf, W_int1, b_int1, bufB, 16,  256, 0,    0,    0);
    // int2 : B (128) -> A (B,64,T), groups=8
    gconv_wmma<16, 0, 8, true , __bf16, __bf16, __bf16><<<dim3(NT,  8, 16), blk, 0, stream>>>(
        bufB, nobf, W_int2, b_int2, bufA,  8,  128, 0,    0,    0);
    // int3 : A (64) -> B (B,32,T), groups=4
    gconv_wmma<16, 0, 8, true , __bf16, __bf16, __bf16><<<dim3(NT,  4, 16), blk, 0, stream>>>(
        bufA, nobf, W_int3, b_int3, bufB,  4,   64, 0,    0,    0);
    // int4 : B (32) -> A (B,16,T), groups=2
    gconv_wmma<16, 0, 8, true , __bf16, __bf16, __bf16><<<dim3(NT,  2, 16), blk, 0, stream>>>(
        bufB, nobf, W_int4, b_int4, bufA,  2,   32, 0,    0,    0);
    // root : A (16) -> out (B,1,T) f32, groups=1, no activation
    gconv_wmma<16, 0, 1, false, __bf16, __bf16, float ><<<dim3(NT,  1, 16), blk, 0, stream>>>(
        bufA, nobf, W_root, b_root, (float*)d_out, 1, 16, 0, 0, 0);
}
